// TemporalDecomposedAttention_72189810311511
// MI455X (gfx1250) — matrix operand order repaired and checked by reading, not compiled, not run on hardware
//
#include <hip/hip_runtime.h>

// ---------------- problem constants ----------------
constexpr int Bb   = 16;
constexpr int Tt   = 9;
constexpr int Nn   = 1024;
constexpr int DIMc = 768;
constexpr int Hh   = 8;
constexpr int Dd   = 96;               // DIM / HEADS
constexpr int K3c  = 3 * DIMc;         // 2304
constexpr int Mrows = Bb * Tt * Nn;    // 147456

// ---------------- types ----------------
typedef __attribute__((ext_vector_type(16))) __bf16 v16bf;
typedef __attribute__((ext_vector_type(8)))  float  v8f;

struct Frag {
    union {
        uint4 q[2];   // two 16-byte chunks = 16 bf16
        v16bf v;
    };
};

// fp32 -> bf16 bits, round-to-nearest-even
static __device__ __forceinline__ unsigned short f2bf(float f) {
    union { float f; unsigned int u; } c; c.f = f;
    unsigned int u = c.u;
    unsigned int r = (u + 0x7FFFu + ((u >> 16) & 1u)) >> 16;
    return (unsigned short)r;
}
static __device__ __forceinline__ float bf2f(unsigned short h) {
    union { unsigned int u; float f; } c; c.u = ((unsigned int)h) << 16;
    return c.f;
}

// ---------------- conversion kernels ----------------
__global__ void f32_to_bf16_kernel(const float* __restrict__ src,
                                   unsigned short* __restrict__ dst, size_t n) {
    size_t i = (size_t)blockIdx.x * blockDim.x + threadIdx.x;
    size_t stride = (size_t)gridDim.x * blockDim.x;
    for (; i < n; i += stride) dst[i] = f2bf(src[i]);
}

// dst[n*Kd + k] = bf16(src[k*Ncols + n])   (W stored (K,N) row-major -> (N,K))
__global__ void transpose_f32_to_bf16_kernel(const float* __restrict__ src,
                                             unsigned short* __restrict__ dst,
                                             int Kd, int Ncols) {
    size_t total = (size_t)Kd * Ncols;
    size_t stride = (size_t)gridDim.x * blockDim.x;
    for (size_t i = (size_t)blockIdx.x * blockDim.x + threadIdx.x; i < total; i += stride) {
        int k = (int)(i % Kd);
        int n = (int)(i / Kd);
        dst[i] = f2bf(src[(size_t)k * Ncols + n]);
    }
}

// ---------------- WMMA GEMM: C(MxN) = A(MxK) * B(KxN), B given transposed (N,K) ----
// block tile 128x128, BK=32, 8 waves; wave tile 32x64 = 2x4 of 16x16.
// B tile staged to LDS via global_load_async_to_lds_b128 (double-buffered);
// fragments read back with ds_load_b128. A fragments loaded direct to VGPRs.
// MODE 0: store bf16 to Cbf.  MODE 1: store f32 + bias to Cf.

// issue 2 async 16B copies for this thread: chunk j covers column c=j/4,
// bytes [ (j%4)*16, +16 ) of that column's 64-byte K-run at k0.
static __device__ __forceinline__ void async_copy_btile(
    const unsigned short* __restrict__ Bt, int Kdim, int colBase, int k0,
    uint4* dstStage, int tid)
{
#pragma unroll
    for (int u = 0; u < 2; ++u) {
        const int j    = tid * 2 + u;          // 0..511
        const int c    = j >> 2;               // column within 128-wide tile
        const int part = j & 3;                // 16B chunk within 64B K-run
        const unsigned short* g = Bt + (size_t)(colBase + c) * Kdim + k0 + part * 8;
        unsigned int       ldsoff = (unsigned int)(uintptr_t)(dstStage + j);
        unsigned long long gaddr  = (unsigned long long)(uintptr_t)g;
        asm volatile("global_load_async_to_lds_b128 %0, %1, off"
                     :: "v"(ldsoff), "v"(gaddr) : "memory");
    }
}

template<int MODE>
__global__ __launch_bounds__(256) void wmma_gemm_bf16(
    const unsigned short* __restrict__ A,
    const unsigned short* __restrict__ Bt,
    int Kdim,
    unsigned short* __restrict__ Cbf,
    float* __restrict__ Cf,
    const float* __restrict__ bias,
    int ldc)
{
    __shared__ uint4 bufB[2][512];   // 2 stages x 8KB (128 cols x 64B K-run)

    const int tid  = threadIdx.x;
    const int lane = tid & 31;
    const int wave = tid >> 5;
    const int wm   = wave & 3;       // 4 waves along M
    const int wn   = wave >> 2;      // 2 waves along N
    const int r    = lane & 15;
    const int hi   = lane >> 4;      // 0: lanes 0-15, 1: lanes 16-31

    const int row0     = blockIdx.x * 128 + wm * 32;   // wave M base
    const int colBase  = blockIdx.y * 128;             // block N base
    const int col0     = colBase + wn * 64;            // wave N base

    v8f acc[2][4];
#pragma unroll
    for (int mi = 0; mi < 2; ++mi)
#pragma unroll
        for (int ni = 0; ni < 4; ++ni)
            acc[mi][ni] = (v8f)0.0f;

    // prologue: stage first B tile into buffer 0
    async_copy_btile(Bt, Kdim, colBase, 0, &bufB[0][0], tid);

    int cur = 0;
    for (int k0 = 0; k0 < Kdim; k0 += 32) {
        // own-wave async copies into bufB[cur] complete, then barrier so the
        // whole tile (written by all waves) is visible.
        asm volatile("s_wait_asynccnt 0x0" ::: "memory");
        __syncthreads();

        // kick off next k-step's B tile into the other stage
        if (k0 + 32 < Kdim)
            async_copy_btile(Bt, Kdim, colBase, k0 + 32, &bufB[cur ^ 1][0], tid);

        // speculative prefetch of next A rows (dropped silently if OOB)
        __builtin_prefetch(A + (size_t)(row0 + r) * Kdim + k0 + 32, 0, 1);

        // A fragments (direct): lane<16: K = {0..7, 16..23}; lane>=16: +8
        Frag afr[2];
#pragma unroll
        for (int mi = 0; mi < 2; ++mi) {
            const int rowA = row0 + mi * 16 + r;
            const uint4* pa = reinterpret_cast<const uint4*>(
                A + (size_t)rowA * Kdim + k0 + hi * 8);
            afr[mi].q[0] = pa[0];     // K = klo .. klo+7
            afr[mi].q[1] = pa[2];     // K = klo+16 .. klo+23
        }

        // B fragments from LDS: lanes 0-15 hold K={0..7,16..23} of column N=r,
        // lanes 16-31 hold K+8 -> chunks hi and hi+2 of the column run.
        Frag bfr[4];
#pragma unroll
        for (int ni = 0; ni < 4; ++ni) {
            const int colLocal = wn * 64 + ni * 16 + r;           // 0..127
            const uint4* bp = &bufB[cur][colLocal * 4 + hi];
            bfr[ni].q[0] = bp[0];
            bfr[ni].q[1] = bp[2];
        }

#pragma unroll
        for (int mi = 0; mi < 2; ++mi)
#pragma unroll
            for (int ni = 0; ni < 4; ++ni)
                acc[mi][ni] = __builtin_amdgcn_wmma_f32_16x16x32_bf16(
                    false, afr[mi].v, false, bfr[ni].v,
                    (short)0, acc[mi][ni], false, false);

        cur ^= 1;
    }

    // C/D layout: VGPR i -> row = hi*8 + i, col = lane%16
#pragma unroll
    for (int mi = 0; mi < 2; ++mi) {
#pragma unroll
        for (int ni = 0; ni < 4; ++ni) {
            const int col = col0 + ni * 16 + r;
#pragma unroll
            for (int i = 0; i < 8; ++i) {
                const int row = row0 + mi * 16 + hi * 8 + i;
                if (MODE == 0) {
                    Cbf[(size_t)row * ldc + col] = f2bf(acc[mi][ni][i]);
                } else {
                    Cf[(size_t)row * ldc + col] = acc[mi][ni][i] + bias[col];
                }
            }
        }
    }
}

// ---------------- dots[b,h,t,s] = sum_{n,d} q[b,t,n,h,d] * k[b,s,n,h,d] -------
// one 96-thread block per (b,h); thread = d lane; 81 register accumulators.
__global__ __launch_bounds__(96) void dots_kernel(
    const unsigned short* __restrict__ qkv,   // (b,t,n) x 2304 ; q at +0, k at +768
    float* __restrict__ dots)                 // (B*H) x 81 raw sums
{
    const int bh = blockIdx.x;
    const int b  = bh >> 3;
    const int h  = bh & 7;
    const unsigned short* base =
        qkv + (size_t)b * Tt * Nn * K3c + h * Dd + threadIdx.x;

    float acc[81];
#pragma unroll
    for (int i = 0; i < 81; ++i) acc[i] = 0.0f;

    for (int n = 0; n < Nn; ++n) {
        float qv[9], kv[9];
#pragma unroll
        for (int t = 0; t < 9; ++t) {
            size_t off = (size_t)(t * Nn + n) * K3c;
            qv[t] = bf2f(base[off]);
            kv[t] = bf2f(base[off + DIMc]);
        }
#pragma unroll
        for (int t = 0; t < 9; ++t)
#pragma unroll
            for (int s = 0; s < 9; ++s)
                acc[t * 9 + s] = fmaf(qv[t], kv[s], acc[t * 9 + s]);
    }

    __shared__ float red[81];
    if (threadIdx.x < 81) red[threadIdx.x] = 0.0f;
    __syncthreads();
#pragma unroll
    for (int i = 0; i < 81; ++i) atomicAdd(&red[i], acc[i]);
    __syncthreads();
    if (threadIdx.x < 81)
        dots[(size_t)bh * 81 + threadIdx.x] = red[threadIdx.x];
}

// ---------------- softmax over s, mean over h, @W_temp + b_temp ----------------
__global__ __launch_bounds__(128) void softmax_temporal_kernel(
    const float* __restrict__ dots,   // (B*H) x 81
    const float* __restrict__ Wtemp,  // 9x9 row-major
    const float* __restrict__ btemp,  // 9
    float* __restrict__ ta)           // B x 81
{
    const int b = blockIdx.x;
    __shared__ float meanA[81];
    if (threadIdx.x < 81) meanA[threadIdx.x] = 0.0f;
    __syncthreads();

    const float sc = 0.10206207261596575f / 1024.0f;   // HEAD_DIM^-0.5 / N

    if (threadIdx.x < 72) {
        const int h = threadIdx.x / 9;
        const int t = threadIdx.x % 9;
        const float* row = dots + ((size_t)(b * 8 + h) * 9 + t) * 9;
        float v[9], mx = -1e30f;
#pragma unroll
        for (int s = 0; s < 9; ++s) { v[s] = row[s] * sc; mx = fmaxf(mx, v[s]); }
        float e[9], sum = 0.0f;
#pragma unroll
        for (int s = 0; s < 9; ++s) { e[s] = __expf(v[s] - mx); sum += e[s]; }
        const float inv = 1.0f / (sum * 8.0f);   // fold mean over H
#pragma unroll
        for (int s = 0; s < 9; ++s) atomicAdd(&meanA[t * 9 + s], e[s] * inv);
    }
    __syncthreads();

    if (threadIdx.x < 81) {
        const int t = threadIdx.x / 9;
        const int s = threadIdx.x % 9;
        float o = btemp[s];
#pragma unroll
        for (int s2 = 0; s2 < 9; ++s2)
            o = fmaf(meanA[t * 9 + s2], Wtemp[s2 * 9 + s], o);
        ta[(size_t)b * 81 + threadIdx.x] = o;
    }
}

// ---------------- vmix[b,t,n,c] = sum_s ta[b,t,s] * v[b,s,n,c] ----------------
__global__ __launch_bounds__(256) void vmix_kernel(
    const unsigned short* __restrict__ qkv,   // v at col offset 1536
    const float* __restrict__ ta,             // B x 81
    unsigned short* __restrict__ vmix)        // (b,t,n) x 768, bf16
{
    const int b = blockIdx.y;
    __shared__ float taL[81];
    if (threadIdx.x < 81) taL[threadIdx.x] = ta[(size_t)b * 81 + threadIdx.x];
    __syncthreads();

    const int idx = blockIdx.x * 256 + threadIdx.x;   // [0, N*DIM)
    const int n = idx / DIMc;
    const int c = idx % DIMc;

    const unsigned short* vbase =
        qkv + (size_t)b * Tt * Nn * K3c + 2 * DIMc + c;

    float vv[9];
#pragma unroll
    for (int s = 0; s < 9; ++s)
        vv[s] = bf2f(vbase[(size_t)(s * Nn + n) * K3c]);

#pragma unroll
    for (int t = 0; t < 9; ++t) {
        float o = 0.0f;
#pragma unroll
        for (int s = 0; s < 9; ++s) o = fmaf(taL[t * 9 + s], vv[s], o);
        vmix[((size_t)(b * Tt + t) * Nn + n) * DIMc + c] = f2bf(o);
    }
}

// ---------------- launch ----------------
extern "C" void kernel_launch(void* const* d_in, const int* in_sizes, int n_in,
                              void* d_out, int out_size, void* d_ws, size_t ws_size,
                              hipStream_t stream) {
    const float* x     = (const float*)d_in[0];
    const float* Wqkv  = (const float*)d_in[1];
    const float* Wtemp = (const float*)d_in[2];
    const float* btemp = (const float*)d_in[3];
    const float* Wout  = (const float*)d_in[4];
    const float* bout  = (const float*)d_in[5];
    float* out = (float*)d_out;

    // workspace layout (all 16B aligned)
    char* ws = (char*)d_ws;
    constexpr size_t SZ_XBF  = (size_t)Mrows * DIMc * 2;   // 226,492,416
    constexpr size_t SZ_QKV  = (size_t)Mrows * K3c  * 2;   // 679,477,248
    constexpr size_t SZ_VMIX = SZ_XBF;
    constexpr size_t SZ_WQT  = (size_t)K3c * DIMc * 2;
    constexpr size_t SZ_WOT  = (size_t)DIMc * DIMc * 2;
    constexpr size_t SZ_DOTS = (size_t)Bb * Hh * 81 * 4;

    size_t off = 0;
    unsigned short* x_bf   = (unsigned short*)(ws + off); off += SZ_XBF;
    unsigned short* qkv_bf = (unsigned short*)(ws + off); off += SZ_QKV;
    unsigned short* vmix   = (unsigned short*)(ws + off); off += SZ_VMIX;
    unsigned short* WqkvT  = (unsigned short*)(ws + off); off += SZ_WQT;
    unsigned short* WoutT  = (unsigned short*)(ws + off); off += SZ_WOT;
    float*          dotsP  = (float*)(ws + off);          off += SZ_DOTS;
    float*          taP    = (float*)(ws + off);

    // 1) convert x to bf16; transpose-convert weights
    f32_to_bf16_kernel<<<4096, 256, 0, stream>>>(x, x_bf, (size_t)Mrows * DIMc);
    transpose_f32_to_bf16_kernel<<<2048, 256, 0, stream>>>(Wqkv, WqkvT, DIMc, K3c);
    transpose_f32_to_bf16_kernel<<<1024, 256, 0, stream>>>(Wout, WoutT, DIMc, DIMc);

    // 2) qkv = x @ W_qkv   (M=147456, K=768, N=2304) -> bf16
    wmma_gemm_bf16<0><<<dim3(Mrows / 128, K3c / 128), 256, 0, stream>>>(
        x_bf, WqkvT, DIMc, qkv_bf, nullptr, nullptr, K3c);

    // 3) dots[b,h,t,s]
    dots_kernel<<<Bb * Hh, 96, 0, stream>>>(qkv_bf, dotsP);

    // 4) softmax + mean over heads + temporal projection
    softmax_temporal_kernel<<<Bb, 128, 0, stream>>>(dotsP, Wtemp, btemp, taP);

    // 5) vmix = sum_s ta * v  -> bf16
    vmix_kernel<<<dim3((Nn * DIMc) / 256, Bb), 256, 0, stream>>>(qkv_bf, taP, vmix);

    // 6) out = vmix @ W_out + b_out  (M=147456, K=768, N=768) -> f32
    wmma_gemm_bf16<1><<<dim3(Mrows / 128, DIMc / 128), 256, 0, stream>>>(
        vmix, WoutT, DIMc, nullptr, out, bout, DIMc);
}
